// ESN_85091892068407
// MI455X (gfx1250) — matrix operand order repaired and checked by reading, not compiled, not run on hardware
//
#include <hip/hip_runtime.h>
#include <math.h>

// ESN (2-layer leaky reservoir) persistent-scan kernel for gfx1250.
//   - bf16 WMMA 16x16x32, fp32 accumulate (M=16 == batch).
//   - weights repacked once per launch into per-lane WMMA B-fragments (L2-resident).
//   - hidden state broadcast via global buffer -> TDM tensor_load_to_lds into each WG's LDS.
//   - 16 persistent WGs, 2 grid barriers/step, readout fused.

#define T_STEPS  2048
#define RES_N    1024
#define IN_F     64
#define BATCH    16
#define NWG      16
#define NTHREADS 128
#define ALPHA_F  0.5f

typedef __attribute__((ext_vector_type(16))) __bf16    v16bf;
typedef __attribute__((ext_vector_type(8)))  float     v8f;
typedef __attribute__((ext_vector_type(4)))  unsigned  v4u;
typedef __attribute__((ext_vector_type(8)))  int       v8i;
typedef __attribute__((ext_vector_type(4)))  int       v4i;

union FragU { v16bf v; v4u q[2]; };

#if defined(__gfx1250__) && __has_builtin(__builtin_amdgcn_tensor_load_to_lds)
#define USE_TDM 1
#else
#define USE_TDM 0
#endif

__device__ __forceinline__ unsigned short f2bf(float x) {
  unsigned u = __float_as_uint(x);
  unsigned r = u + 0x7FFFu + ((u >> 16) & 1u);   // round-to-nearest-even
  return (unsigned short)(r >> 16);
}
__device__ __forceinline__ float bf2f(unsigned short h) {
  return __uint_as_float(((unsigned)h) << 16);
}

// ---- WMMA fragment loaders -------------------------------------------------
// A (16x32 bf16): lane m=lane&15; lanes<16 hold K{0..7,16..23}, lanes>=16 K{8..15,24..31}.
__device__ __forceinline__ v16bf loadA(const char* base, int rowStride, int kt, int lane) {
  int m = lane & 15;
  const char* p = base + m * rowStride + kt * 64 + ((lane >> 4) << 4);
  FragU f;
  f.q[0] = *(const v4u*)(p);        // K +0..7   (16B)
  f.q[1] = *(const v4u*)(p + 32);   // K +16..23 (16B)
  return f.v;
}
// B (32x16 bf16): pre-repacked; fragment = 32 lanes x 16 bf16 contiguous (1024B).
__device__ __forceinline__ v16bf loadB(const char* frags, int fi, int lane) {
  const char* p = frags + (size_t)fi * 1024 + lane * 32;
  FragU f;
  f.q[0] = *(const v4u*)(p);
  f.q[1] = *(const v4u*)(p + 16);
  return f.v;
}

__device__ __forceinline__ v8f gemm_tiles(const char* aBase, int aStride, int kt0, int kt1,
                                          const char* Bfrags, int fiBase, int lane, v8f acc) {
  for (int kt = kt0; kt < kt1; ++kt) {
    v16bf a = loadA(aBase, aStride, kt, lane);
    v16bf b = loadB(Bfrags, fiBase + kt, lane);
    acc = __builtin_amdgcn_wmma_f32_16x16x32_bf16(false, a, false, b, (short)0, acc,
                                                  false, false);
  }
  return acc;
}

// ---- global -> LDS hidden-state refill (TDM per wave: 4 rows x 2048B) ------
__device__ __forceinline__ void refill(unsigned short* dstLds, const unsigned short* src,
                                       int wave) {
#if USE_TDM
  unsigned ldsAddr = (unsigned)(size_t)(void*)(dstLds + (size_t)wave * 4 * RES_N);
  unsigned long long ga = (unsigned long long)(size_t)(const void*)(src + (size_t)wave * 4 * RES_N);
  // D# group0: count=1, lds_addr, global_addr, type=2
  v4u g0 = { 1u, ldsAddr, (unsigned)ga, (unsigned)(ga >> 32) | (2u << 30) };
  // group1: data_size=3 (8B units), tensor_dim0=256 (=2048B row), tensor_dim1=4,
  //         tile_dim0=256, tile_dim1=4, tensor_dim0_stride=256
  v8i g1 = { (int)0x30000u, (int)(256u << 16), (int)(4u << 16), (int)(256u << 16),
             4, 256, 0, 0 };
  v4i g2 = { 0, 0, 0, 0 };
  v4i g3 = { 0, 0, 0, 0 };
  v8i g4 = { 0, 0, 0, 0, 0, 0, 0, 0 };
  __builtin_amdgcn_tensor_load_to_lds(g0, g1, g2, g3, g4, 0);
  __builtin_amdgcn_s_wait_tensorcnt(0);
  __syncthreads();   // all 4 waves' TDM tiles landed
#else
  for (int i = threadIdx.x; i < (BATCH * RES_N) / 8; i += NTHREADS)
    ((v4u*)dstLds)[i] = ((const v4u*)src)[i];
  __syncthreads();
#endif
}

// ---- monotonic grid barrier (counter zeroed each launch) -------------------
__device__ __forceinline__ void grid_barrier(unsigned* ctr, unsigned& phase) {
  phase += NWG;
  __threadfence();
  __syncthreads();
  if (threadIdx.x == 0) {
    __hip_atomic_fetch_add(ctr, 1u, __ATOMIC_RELEASE, __HIP_MEMORY_SCOPE_AGENT);
    while (__hip_atomic_load(ctr, __ATOMIC_ACQUIRE, __HIP_MEMORY_SCOPE_AGENT) < phase) {
#if defined(__HIP_DEVICE_COMPILE__)
      __builtin_amdgcn_s_sleep(2);
#endif
    }
  }
  __syncthreads();
  __threadfence();
}

// ---- setup kernels ---------------------------------------------------------
__global__ void zero_ctr(unsigned* c) {
  if (threadIdx.x == 0 && blockIdx.x == 0) *c = 0u;
}

// x[B][T][64] fp32 -> Xb[t][m][64] bf16 (time-major A panels)
__global__ void repack_x(const float* __restrict__ x, unsigned short* __restrict__ Xb) {
  int id = blockIdx.x * blockDim.x + threadIdx.x;
  if (id >= T_STEPS * BATCH * IN_F) return;
  int k = id & (IN_F - 1);
  int m = (id >> 6) & (BATCH - 1);
  int t = id >> 10;
  Xb[id] = f2bf(x[((size_t)m * T_STEPS + t) * IN_F + k]);
}

// W[K][N] fp32 row-major -> bf16 WMMA B-fragments.
// fragment fi = nt*(K/32)+kt ; lane: n = nt*16+(lane&15), K = kt*32 + (lane>=16?8:0) + i + (i>=8?8:0)
__global__ void repack_w(const float* __restrict__ W, unsigned short* __restrict__ dst,
                         int K, int N) {
  int id = blockIdx.x * blockDim.x + threadIdx.x;
  int KT = K >> 5;
  int total = (N >> 4) * KT * 32;
  if (id >= total) return;
  int lane = id & 31;
  int fi = id >> 5;
  int kt = fi % KT;
  int nt = fi / KT;
  int n = nt * 16 + (lane & 15);
  int kbase = kt * 32 + ((lane >> 4) << 3);
  unsigned short* o = dst + (size_t)id * 16;
#pragma unroll
  for (int i = 0; i < 16; ++i)
    o[i] = f2bf(W[(size_t)(kbase + i + ((i >> 3) << 3)) * N + n]);
}

// ---- persistent scan kernel ------------------------------------------------
__global__ void __launch_bounds__(NTHREADS) esn_scan(
    const unsigned short* __restrict__ Xb,
    const char* __restrict__ WbIn0, const char* __restrict__ WbHh0,
    const char* __restrict__ WbIn1, const char* __restrict__ WbHh1,
    const char* __restrict__ WbRo,
    const float* __restrict__ b0, const float* __restrict__ b1,
    unsigned short* __restrict__ H0g, unsigned short* __restrict__ H1g,
    unsigned* __restrict__ ctr,
    float* __restrict__ out) {
  __shared__ __attribute__((aligned(16))) unsigned short sh_h0[BATCH * RES_N];  // 32KB
  __shared__ __attribute__((aligned(16))) unsigned short sh_h1[BATCH * RES_N];  // 32KB
  __shared__ __attribute__((aligned(16))) float red[4 * 256];                   // 4KB

  const int tid  = threadIdx.x;
  const int lane = tid & 31;
  const int wave = tid >> 5;
  const int wg   = blockIdx.x;
  const int nt   = wg * 4 + wave;             // N-tile 0..63 owned by this wave
  const int grp  = lane >> 4;                 // 0/1 half-lane group
  const int col  = nt * 16 + (lane & 15);     // global output column
  const float bv0 = b0[col];
  const float bv1 = b1[col];

  for (int i = tid; i < BATCH * RES_N; i += NTHREADS) { sh_h0[i] = 0; sh_h1[i] = 0; }
  __syncthreads();

  const char* sh0c = (const char*)sh_h0;
  const char* sh1c = (const char*)sh_h1;
  unsigned phase = 0;

  for (int t = 0; t < T_STEPS; ++t) {
    // ---------------- layer 0: h0' = (1-a)h0 + a*tanh(x_t@Win0 + h0@Whh0 + b0)
    {
      v8f acc = {0.f, 0.f, 0.f, 0.f, 0.f, 0.f, 0.f, 0.f};
      const char* xb = (const char*)(Xb + (size_t)t * BATCH * IN_F);
      acc = gemm_tiles(xb, IN_F * 2, 0, IN_F / 32, WbIn0, nt * (IN_F / 32), lane, acc);
      acc = gemm_tiles(sh0c, RES_N * 2, 0, RES_N / 32, WbHh0, nt * (RES_N / 32), lane, acc);
#pragma unroll
      for (int r = 0; r < 8; ++r) {
        int m = r + grp * 8;
        float hold = bf2f(sh_h0[m * RES_N + col]);
        float hn = (1.f - ALPHA_F) * hold + ALPHA_F * tanhf(acc[r] + bv0);
        H0g[m * RES_N + col] = f2bf(hn);
      }
    }
    grid_barrier(ctr, phase);
    refill(sh_h0, H0g, wave);   // sh_h0 := h0_new (broadcast to all WGs)

    // ---------------- layer 1: h1' = (1-a)h1 + a*tanh(h0'@Win1 + h1@Whh1 + b1)
    {
      v8f acc = {0.f, 0.f, 0.f, 0.f, 0.f, 0.f, 0.f, 0.f};
      acc = gemm_tiles(sh0c, RES_N * 2, 0, RES_N / 32, WbIn1, nt * (RES_N / 32), lane, acc);
      acc = gemm_tiles(sh1c, RES_N * 2, 0, RES_N / 32, WbHh1, nt * (RES_N / 32), lane, acc);
#pragma unroll
      for (int r = 0; r < 8; ++r) {
        int m = r + grp * 8;
        float hold = bf2f(sh_h1[m * RES_N + col]);
        float hn = (1.f - ALPHA_F) * hold + ALPHA_F * tanhf(acc[r] + bv1);
        H1g[m * RES_N + col] = f2bf(hn);
      }
    }
    grid_barrier(ctr, phase);
    refill(sh_h1, H1g, wave);   // sh_h1 := h1_new

    // ---------------- fused readout: out_t = [h0'; h1'] @ readout  (WGs 0..3)
    if (wg < 4) {
      v8f racc = {0.f, 0.f, 0.f, 0.f, 0.f, 0.f, 0.f, 0.f};
      const char* ab = (wave < 2) ? sh0c : sh1c;
      int ktb = (wave & 1) * 16;                       // kt range within its h-buffer
      racc = gemm_tiles(ab, RES_N * 2, ktb, ktb + 16,
                        WbRo, wg * 64 + wave * 16 - ktb, lane, racc);
#pragma unroll
      for (int r = 0; r < 8; ++r) red[wave * 256 + r * 32 + lane] = racc[r];
      __syncthreads();
      for (int e = tid; e < 256; e += NTHREADS) {
        float s = red[e] + red[256 + e] + red[512 + e] + red[768 + e];
        int r2 = e >> 5, le = e & 31;
        int m = r2 + ((le >> 4) << 3);
        int n = wg * 16 + (le & 15);
        out[((size_t)m * T_STEPS + t) * IN_F + n] = s;   // out[b][t][n]
      }
    }
  }
}

// ---- host entry ------------------------------------------------------------
extern "C" void kernel_launch(void* const* d_in, const int* in_sizes, int n_in,
                              void* d_out, int out_size, void* d_ws, size_t ws_size,
                              hipStream_t stream) {
  const float* x    = (const float*)d_in[0];
  const float* Win0 = (const float*)d_in[1];
  const float* Whh0 = (const float*)d_in[2];
  const float* b0   = (const float*)d_in[3];
  const float* Win1 = (const float*)d_in[4];
  const float* Whh1 = (const float*)d_in[5];
  const float* b1   = (const float*)d_in[6];
  const float* ro   = (const float*)d_in[7];
  float* out = (float*)d_out;

  char* ws = (char*)d_ws;
  size_t off = 0;
  unsigned* ctr = (unsigned*)ws;                        off = 256;
  unsigned short* H0 = (unsigned short*)(ws + off);     off += (size_t)BATCH * RES_N * 2;
  unsigned short* H1 = (unsigned short*)(ws + off);     off += (size_t)BATCH * RES_N * 2;
  unsigned short* Xb = (unsigned short*)(ws + off);     off += (size_t)T_STEPS * BATCH * IN_F * 2;
  char* WbIn0 = ws + off;  off += (size_t)(RES_N / 16) * (IN_F / 32) * 1024;
  char* WbHh0 = ws + off;  off += (size_t)(RES_N / 16) * (RES_N / 32) * 1024;
  char* WbIn1 = ws + off;  off += (size_t)(RES_N / 16) * (RES_N / 32) * 1024;
  char* WbHh1 = ws + off;  off += (size_t)(RES_N / 16) * (RES_N / 32) * 1024;
  char* WbRo  = ws + off;  off += (size_t)(IN_F / 16) * (2 * RES_N / 32) * 1024;
  (void)ws_size; (void)in_sizes; (void)n_in; (void)out_size;

  zero_ctr<<<1, 64, 0, stream>>>(ctr);
  {
    int total = T_STEPS * BATCH * IN_F;
    repack_x<<<(total + 255) / 256, 256, 0, stream>>>(x, Xb);
  }
  {
    int total;
    total = (RES_N / 16) * (IN_F / 32) * 32;
    repack_w<<<(total + 255) / 256, 256, 0, stream>>>(Win0, (unsigned short*)WbIn0, IN_F, RES_N);
    total = (RES_N / 16) * (RES_N / 32) * 32;
    repack_w<<<(total + 255) / 256, 256, 0, stream>>>(Whh0, (unsigned short*)WbHh0, RES_N, RES_N);
    repack_w<<<(total + 255) / 256, 256, 0, stream>>>(Win1, (unsigned short*)WbIn1, RES_N, RES_N);
    repack_w<<<(total + 255) / 256, 256, 0, stream>>>(Whh1, (unsigned short*)WbHh1, RES_N, RES_N);
    total = (IN_F / 16) * (2 * RES_N / 32) * 32;
    repack_w<<<(total + 255) / 256, 256, 0, stream>>>(ro, (unsigned short*)WbRo, 2 * RES_N, IN_F);
  }
  esn_scan<<<NWG, NTHREADS, 0, stream>>>(Xb, WbIn0, WbHh0, WbIn1, WbHh1, WbRo,
                                         b0, b1, H0, H1, ctr, out);
}
